// TransformerBlock_10531259810703
// MI455X (gfx1250) — matrix operand order repaired and checked by reading, not compiled, not run on hardware
//
#include <hip/hip_runtime.h>
#include <hip/hip_bf16.h>
#include <cstdint>
#include <cstddef>

// ---------------- CDNA5 WMMA types ----------------
typedef __attribute__((ext_vector_type(16))) _Float16 v16h;
typedef __attribute__((ext_vector_type(8)))  float    v8f;

union Frag16 {          // one 16x32(A) or 32x16(B) f16 fragment per lane
    v16h     v;
    uint4    q[2];
    _Float16 h[16];
};

// Problem constants (from reference)
#define BB   2
#define TT   2048
#define HH   1024
#define NHH  16
#define DKK  64
#define DFFF 4096
#define EE   8
#define NN   (BB*TT)        // 4096 tokens
#define CAPN 1280           // int(1.25 * (N*2/E))

// ---------------- CDNA5 async global->LDS path ----------------
#if defined(__has_builtin)
#  if __has_builtin(__builtin_amdgcn_global_load_async_to_lds_b128)
#    define HAVE_ASYNC_B128 1
#  endif
#  if __has_builtin(__builtin_amdgcn_s_wait_asynccnt)
#    define HAVE_WAIT_ASYNC 1
#  endif
#endif

typedef int vint4_t __attribute__((vector_size(16)));

__device__ __forceinline__ void async_copy16(const _Float16* g, _Float16* l)
{
#if defined(HAVE_ASYNC_B128)
    __builtin_amdgcn_global_load_async_to_lds_b128(
        (__attribute__((address_space(1))) vint4_t*)g,
        (__attribute__((address_space(3))) vint4_t*)l, 0, 0);
#else
    *(uint4*)l = *(const uint4*)g;
#endif
}
__device__ __forceinline__ void wait_async()
{
#if defined(HAVE_ASYNC_B128)
#  if defined(HAVE_WAIT_ASYNC)
    __builtin_amdgcn_s_wait_asynccnt(0);
#  else
    asm volatile("s_wait_asynccnt 0" ::: "memory");
#  endif
#endif
}

// =====================================================================
// Tiled WMMA GEMM: C[M,N] = A[M,K](f16,row) @ B[K,N](f16,row)
// block tile 128x128, K-step 32, 8 waves -> each wave 32(M)x64(N).
// Double-buffered LDS; A tiles staged with GLOBAL_LOAD_ASYNC_TO_LDS_B128,
// B tiles transposed through VGPRs (needed for the [N][K] fragment layout).
// EPI: 0 = store f32, 1 = store f16, 2 = store f32 + residual add
// =====================================================================
template<int EPI>
__global__ __launch_bounds__(256)
void gemm_wmma(const _Float16* __restrict__ A, const _Float16* __restrict__ Bw,
               void* __restrict__ Cout, const float* __restrict__ Res,
               int K, int lda, int ldb, int ldc)
{
    __shared__ _Float16 As[2][128 * 40];   // [m][k], 40-half stride (16B rows)
    __shared__ _Float16 Bs[2][128 * 40];   // [n][k] transposed, same stride

    const int tid  = threadIdx.x;
    const int lane = tid & 31;
    const int wid  = tid >> 5;
    const int wm   = wid & 3;           // 0..3 -> 32-row slice
    const int wn   = wid >> 2;          // 0..1 -> 64-col slice
    const int l16  = lane & 15;
    const int lhi  = lane >> 4;         // 0 or 1
    const int bm   = blockIdx.x * 128;
    const int bn   = blockIdx.y * 128;

    // staging coordinates: 2 vec8 (16B) per thread for each of A and B
    const int va0 = tid * 2,     va1 = tid * 2 + 1;
    const int ar0 = va0 >> 2,    ac0 = (va0 & 3) * 8;
    const int ar1 = va1 >> 2,    ac1 = (va1 & 3) * 8;
    const int bk0 = va0 >> 4,    bn0 = (va0 & 15) * 8;
    const int bk1 = va1 >> 4,    bn1 = (va1 & 15) * 8;

    v8f zero = {};
    v8f acc[2][4];
    #pragma unroll
    for (int i = 0; i < 2; ++i)
        #pragma unroll
        for (int j = 0; j < 4; ++j) acc[i][j] = zero;

    uint4 breg[2];
    auto issueA = [&](int k0, int buf) {
        async_copy16(&A[(size_t)(bm + ar0) * lda + k0 + ac0], &As[buf][ar0 * 40 + ac0]);
        async_copy16(&A[(size_t)(bm + ar1) * lda + k0 + ac1], &As[buf][ar1 * 40 + ac1]);
    };
    auto loadB = [&](int k0) {
        breg[0] = *(const uint4*)&Bw[(size_t)(k0 + bk0) * ldb + bn + bn0];
        breg[1] = *(const uint4*)&Bw[(size_t)(k0 + bk1) * ldb + bn + bn1];
    };
    auto storeB = [&](int buf) {
        const _Float16* t0 = (const _Float16*)&breg[0];
        const _Float16* t1 = (const _Float16*)&breg[1];
        #pragma unroll
        for (int j = 0; j < 8; ++j) Bs[buf][(bn0 + j) * 40 + bk0] = t0[j];
        #pragma unroll
        for (int j = 0; j < 8; ++j) Bs[buf][(bn1 + j) * 40 + bk1] = t1[j];
    };

    const int nk = K >> 5;
    issueA(0, 0);
    loadB(0);
    storeB(0);
    wait_async();
    __syncthreads();

    for (int t = 0; t < nk; ++t) {
        const int cur = t & 1, nxt = cur ^ 1;
        if (t + 1 < nk) {                      // overlap next tile with compute
            issueA((t + 1) << 5, nxt);
            loadB((t + 1) << 5);
        }
        if (t + 2 < nk) {                      // L2 prefetch for the B stream
            __builtin_prefetch(&Bw[(size_t)(((t + 2) << 5) + bk0) * ldb + bn + bn0], 0, 1);
        }

        Frag16 a[2], b[4];
        const int koffA = lhi ? 8 : 0;         // A: K = {0..7,16..23} (+8 hi half)
        #pragma unroll
        for (int fm = 0; fm < 2; ++fm) {
            int row = wm * 32 + fm * 16 + l16;
            a[fm].q[0] = *(const uint4*)&As[cur][row * 40 + koffA];
            a[fm].q[1] = *(const uint4*)&As[cur][row * 40 + koffA + 16];
        }
        const int koffB = lhi ? 8 : 0;         // B: K = {0..7,16..23} (+8 hi half)
        #pragma unroll
        for (int fn = 0; fn < 4; ++fn) {
            int col = wn * 64 + fn * 16 + l16;
            b[fn].q[0] = *(const uint4*)&Bs[cur][col * 40 + koffB];
            b[fn].q[1] = *(const uint4*)&Bs[cur][col * 40 + koffB + 16];
        }
        #pragma unroll
        for (int fm = 0; fm < 2; ++fm)
            #pragma unroll
            for (int fn = 0; fn < 4; ++fn)
                acc[fm][fn] = __builtin_amdgcn_wmma_f32_16x16x32_f16(
                    false, a[fm].v, false, b[fn].v, (short)0, acc[fm][fn],
                    false, false);

        if (t + 1 < nk) storeB(nxt);
        wait_async();
        __syncthreads();
    }

    // epilogue: C layout -> lane holds col l16, rows r + 8*lhi
    #pragma unroll
    for (int fm = 0; fm < 2; ++fm)
        #pragma unroll
        for (int fn = 0; fn < 4; ++fn)
            #pragma unroll
            for (int r = 0; r < 8; ++r) {
                int row = bm + wm * 32 + fm * 16 + r + lhi * 8;
                int col = bn + wn * 64 + fn * 16 + l16;
                float vv = acc[fm][fn][r];
                if (EPI == 0)
                    ((float*)Cout)[(size_t)row * ldc + col] = vv;
                else if (EPI == 1)
                    ((_Float16*)Cout)[(size_t)row * ldc + col] = (_Float16)vv;
                else
                    ((float*)Cout)[(size_t)row * ldc + col] =
                        vv + Res[(size_t)row * ldc + col];
            }
}

// =====================================================================
// LayerNorm (H=1024) -> f32 + f16 outputs
// =====================================================================
__global__ __launch_bounds__(256)
void ln_kernel(const float* __restrict__ X, const float* __restrict__ g,
               const float* __restrict__ b, float* __restrict__ outf,
               _Float16* __restrict__ outh)
{
    __shared__ float red[8];
    __shared__ float s_mu, s_rstd;
    const int H = HH;
    const int row = blockIdx.x, tid = threadIdx.x;
    const float* xr = X + (size_t)row * H;

    float s = 0.f;
    for (int c = tid; c < H; c += 256) s += xr[c];
    for (int o = 16; o; o >>= 1) s += __shfl_xor(s, o, 32);
    if ((tid & 31) == 0) red[tid >> 5] = s;
    __syncthreads();
    if (tid == 0) { float t = 0; for (int i = 0; i < 8; ++i) t += red[i]; s_mu = t / H; }
    __syncthreads();
    float mu = s_mu;

    float v = 0.f;
    for (int c = tid; c < H; c += 256) { float d = xr[c] - mu; v += d * d; }
    for (int o = 16; o; o >>= 1) v += __shfl_xor(v, o, 32);
    if ((tid & 31) == 0) red[tid >> 5] = v;
    __syncthreads();
    if (tid == 0) { float t = 0; for (int i = 0; i < 8; ++i) t += red[i]; s_rstd = rsqrtf(t / H + 1e-5f); }
    __syncthreads();
    float rs = s_rstd;

    for (int c = tid; c < H; c += 256) {
        float y = (xr[c] - mu) * rs * g[c] + b[c];
        outf[(size_t)row * H + c] = y;
        outh[(size_t)row * H + c] = (_Float16)y;
    }
}

// =====================================================================
// Causal attention, one (b,h) x 32 query rows per block.
// Scores for the causal key range live in dynamic LDS; V chunks are staged
// cooperatively (transposed) so P@V B-fragments are two ds_load_b128s.
// =====================================================================
#define SROW 2056
#define VSTRIDE 40
#define ATT_SHM (32 * SROW * sizeof(float) + 64 * VSTRIDE * sizeof(_Float16))

__global__ __launch_bounds__(256)
void att_kernel(const _Float16* __restrict__ qkv, _Float16* __restrict__ o16)
{
    extern __shared__ float sc[];
    _Float16* vbuf = (_Float16*)(sc + 32 * SROW);   // [64 d][VSTRIDE] halves

    const int H3 = 3 * HH;
    const int bh = blockIdx.x;
    const int bb = bh >> 4, hh = bh & 15;
    const int qbase = blockIdx.y * 32;
    const int tid = threadIdx.x, lane = tid & 31, wid = tid >> 5;
    const int l16 = lane & 15, lhi = lane >> 4;
    const _Float16* base = qkv + (size_t)bb * TT * H3;
    const int ntiles = (qbase + 32) >> 4;        // always even
    const int ncols  = ntiles * 16;

    // ---- preload Q fragments (reused across every key tile) ----
    Frag16 aq[2][2];
    #pragma unroll
    for (int qt = 0; qt < 2; ++qt)
        #pragma unroll
        for (int kc = 0; kc < 2; ++kc) {
            int qtok = qbase + qt * 16 + l16;
            int koffA = kc * 32 + (lhi ? 8 : 0);
            const _Float16* ap = base + (size_t)qtok * H3 + hh * DKK + koffA;
            aq[qt][kc].q[0] = *(const uint4*)ap;
            aq[qt][kc].q[1] = *(const uint4*)(ap + 16);
        }

    // ---- phase 1: scores S (scaled, masked) into LDS ----
    for (int idx = wid; idx < 2 * ntiles; idx += 8) {
        int qt = idx & 1, kt = idx >> 1;
        v8f c = {};
        #pragma unroll
        for (int kc = 0; kc < 2; ++kc) {         // d = kc*32 .. +31
            Frag16 bf;
            int ktok = kt * 16 + l16;
            int koffB = kc * 32 + (lhi ? 8 : 0);
            const _Float16* bp = base + (size_t)ktok * H3 + HH + hh * DKK + koffB;
            bf.q[0] = *(const uint4*)bp;
            bf.q[1] = *(const uint4*)(bp + 16);
            c = __builtin_amdgcn_wmma_f32_16x16x32_f16(false, aq[qt][kc].v, false,
                                                       bf.v, (short)0, c, false, false);
        }
        #pragma unroll
        for (int r = 0; r < 8; ++r) {
            int ql = qt * 16 + r + lhi * 8;      // local q row 0..31
            int kg = kt * 16 + l16;              // global k col
            float vv = (kg <= qbase + ql) ? c[r] * 0.125f : -1.0e9f;
            sc[ql * SROW + kg] = vv;
        }
    }
    __syncthreads();

    // ---- phase 2: row softmax (wave wid owns rows 4*wid..4*wid+3) ----
    for (int rr = 0; rr < 4; ++rr) {
        float* srow = sc + (wid * 4 + rr) * SROW;
        float m = -3.0e38f;
        for (int c0 = lane; c0 < ncols; c0 += 32) m = fmaxf(m, srow[c0]);
        for (int o = 16; o; o >>= 1) m = fmaxf(m, __shfl_xor(m, o, 32));
        float s = 0.f;
        for (int c0 = lane; c0 < ncols; c0 += 32) {
            float e = __expf(srow[c0] - m);
            srow[c0] = e; s += e;
        }
        for (int o = 16; o; o >>= 1) s += __shfl_xor(s, o, 32);
        float inv = 1.0f / s;
        for (int c0 = lane; c0 < ncols; c0 += 32) srow[c0] *= inv;
    }
    __syncthreads();

    // ---- phase 3: O = P @ V (8 waves -> 8 16x16 output tiles) ----
    const int qt = wid & 1, dt = wid >> 1;
    const int krow = tid >> 3;            // 0..31 : k row staged by this thread
    const int c8   = (tid & 7) * 8;       // 0..56 : d offset staged (8 halves)
    v8f c = {};
    for (int kc = 0; kc < (ncols >> 5); ++kc) {
        __syncthreads();                  // previous chunk's reads complete
        {   // cooperative transpose-stage of V[kc*32..+31][0..63] -> vbuf[d][k]
            int ktok = kc * 32 + krow;
            uint4 tv = *(const uint4*)&base[(size_t)ktok * H3 + 2 * HH + hh * DKK + c8];
            const _Float16* th = (const _Float16*)&tv;
            #pragma unroll
            for (int j = 0; j < 8; ++j) vbuf[(c8 + j) * VSTRIDE + krow] = th[j];
        }
        __syncthreads();

        Frag16 a;                         // P fragment from probability LDS
        int rowl = qt * 16 + l16;
        int kb = kc * 32 + (lhi ? 8 : 0);
        #pragma unroll
        for (int e = 0; e < 16; ++e) {
            int k = kb + ((e < 8) ? e : e + 8);
            a.h[e] = (_Float16)sc[rowl * SROW + k];
        }
        Frag16 bf;                        // V fragment: two ds_load_b128
        int dcol = dt * 16 + l16;
        int koffB = lhi ? 8 : 0;
        bf.q[0] = *(const uint4*)&vbuf[dcol * VSTRIDE + koffB];
        bf.q[1] = *(const uint4*)&vbuf[dcol * VSTRIDE + koffB + 16];
        c = __builtin_amdgcn_wmma_f32_16x16x32_f16(false, a.v, false, bf.v,
                                                   (short)0, c, false, false);
    }
    #pragma unroll
    for (int r = 0; r < 8; ++r) {
        int tok = qbase + qt * 16 + r + lhi * 8;
        o16[((size_t)(bb * TT + tok)) * HH + hh * DKK + dt * 16 + l16] = (_Float16)c[r];
    }
}

// =====================================================================
// Router: logits = xn2 @ Wr + threefry-normal*0.01, softmax, top-2
// =====================================================================
__device__ inline unsigned rotl32(unsigned x, unsigned n) {
    return (x << n) | (x >> (32 - n));
}
__device__ inline void threefry2x32(unsigned k0, unsigned k1,
                                    unsigned c0, unsigned c1,
                                    unsigned& o0, unsigned& o1)
{
    const unsigned R[8] = {13, 15, 26, 6, 17, 29, 16, 24};
    unsigned ks[3] = {k0, k1, k0 ^ k1 ^ 0x1BD11BDAu};
    unsigned x0 = c0 + k0, x1 = c1 + k1;
    #pragma unroll
    for (int i = 0; i < 5; ++i) {
        #pragma unroll
        for (int r = 0; r < 4; ++r) {
            x0 += x1; x1 = rotl32(x1, R[(i & 1) * 4 + r]); x1 ^= x0;
        }
        x0 += ks[(i + 1) % 3];
        x1 += ks[(i + 2) % 3] + (unsigned)(i + 1);
    }
    o0 = x0; o1 = x1;
}
__device__ inline float jax_normal_bits(unsigned bits) {
    float f = __uint_as_float(0x3f800000u | (bits >> 9));   // [1,2)
    float u = 2.0f * (f - 1.0f) - 1.0f;                     // (-1,1)
    if (u <= -1.0f) u = -0.999999f;
    return 1.41421356f * erfinvf(u);
}

__global__ __launch_bounds__(256)
void router_kernel(const float* __restrict__ xn2, const float* __restrict__ Wr,
                   float* __restrict__ gate, int* __restrict__ idx1,
                   int* __restrict__ idx2, float* __restrict__ val1,
                   float* __restrict__ val2)
{
    int n = blockIdx.x * blockDim.x + threadIdx.x;
    if (n >= NN) return;
    float acc[EE] = {0, 0, 0, 0, 0, 0, 0, 0};
    const float* xr = xn2 + (size_t)n * HH;
    for (int h = 0; h < HH; ++h) {
        float xv = xr[h];
        #pragma unroll
        for (int e = 0; e < EE; ++e) acc[e] += xv * Wr[h * EE + e];
    }
    const unsigned half = (NN * EE) / 2;
    #pragma unroll
    for (int e = 0; e < EE; ++e) {
        unsigned idx = (unsigned)(n * EE + e), o0, o1;
        if (idx < half) { threefry2x32(0u, 42u, idx, idx + half, o0, o1); }
        else            { threefry2x32(0u, 42u, idx - half, idx, o0, o1); o0 = o1; }
        acc[e] += 0.01f * jax_normal_bits(o0);
    }
    float m = acc[0];
    #pragma unroll
    for (int e = 1; e < EE; ++e) m = fmaxf(m, acc[e]);
    float s = 0.f;
    #pragma unroll
    for (int e = 0; e < EE; ++e) { acc[e] = __expf(acc[e] - m); s += acc[e]; }
    float inv = 1.0f / s;
    #pragma unroll
    for (int e = 0; e < EE; ++e) {
        acc[e] *= inv;
        gate[(size_t)n * EE + e] = acc[e];
    }
    int b1 = 0; float v1 = acc[0];
    #pragma unroll
    for (int e = 1; e < EE; ++e) if (acc[e] > v1) { v1 = acc[e]; b1 = e; }
    int b2 = -1; float v2 = -1.f;
    #pragma unroll
    for (int e = 0; e < EE; ++e)
        if (e != b1 && acc[e] > v2) { v2 = acc[e]; b2 = e; }
    idx1[n] = b1; idx2[n] = b2; val1[n] = v1; val2[n] = v2;
}

// =====================================================================
// Capacity dispatch (sequential cumsum, single block)
// =====================================================================
__global__ __launch_bounds__(256)
void dispatch_kernel(const int* __restrict__ idx1, const int* __restrict__ idx2,
                     int* __restrict__ dest, int* __restrict__ src_of_slot,
                     float* __restrict__ used)
{
    int tid = threadIdx.x;
    for (int i = tid; i < EE * CAPN; i += 256) src_of_slot[i] = -1;
    __syncthreads();
    if (tid == 0) {
        int cnt[EE] = {0}, usd[EE] = {0};
        for (int i = 0; i < 2 * NN; ++i) {
            int tok = (i < NN) ? i : i - NN;
            int e   = (i < NN) ? idx1[tok] : idx2[tok];
            int pos = cnt[e]++;
            if (pos < CAPN) {
                dest[i] = e * CAPN + pos;
                src_of_slot[e * CAPN + pos] = tok;
                usd[e]++;
            } else dest[i] = -1;
        }
        for (int e = 0; e < EE; ++e) used[e] = (float)usd[e];
    }
}

__global__ __launch_bounds__(256)
void gather_kernel(const float* __restrict__ xn2, const int* __restrict__ src_of_slot,
                   _Float16* __restrict__ eb)
{
    int slot = blockIdx.x;
    int s = src_of_slot[slot];
    for (int c = threadIdx.x; c < HH; c += 256)
        eb[(size_t)slot * HH + c] =
            (s >= 0) ? (_Float16)xn2[(size_t)s * HH + c] : (_Float16)0.f;
}

__global__ __launch_bounds__(256)
void swiglu_kernel(const _Float16* __restrict__ hpre, _Float16* __restrict__ act)
{
    size_t row = blockIdx.x;
    for (int f = threadIdx.x; f < DFFF; f += 256) {
        float a = (float)hpre[row * (2 * DFFF) + f];
        float g = (float)hpre[row * (2 * DFFF) + DFFF + f];
        float s = a / (1.0f + __expf(-a));
        act[row * DFFF + f] = (_Float16)(s * g);
    }
}

__global__ __launch_bounds__(256)
void combine_kernel(const float* __restrict__ x1, const float* __restrict__ eo,
                    const int* __restrict__ dest, const float* __restrict__ val1,
                    const float* __restrict__ val2, float* __restrict__ out)
{
    int tok = blockIdx.x;
    int d1 = dest[tok], d2 = dest[tok + NN];
    float w1 = (d1 >= 0) ? val1[tok] : 0.f;
    float w2 = (d2 >= 0) ? val2[tok] : 0.f;
    const float* e1 = eo + (size_t)((d1 >= 0) ? d1 : 0) * HH;
    const float* e2 = eo + (size_t)((d2 >= 0) ? d2 : 0) * HH;
    for (int c = threadIdx.x; c < HH; c += 256)
        out[(size_t)tok * HH + c] =
            x1[(size_t)tok * HH + c] + w1 * e1[c] + w2 * e2[c];
}

__global__ __launch_bounds__(256)
void gatesum_kernel(const float* __restrict__ gate, float* __restrict__ prob)
{
    __shared__ float red[8];
    int e = blockIdx.x, tid = threadIdx.x;
    float s = 0.f;
    for (int n = tid; n < NN; n += 256) s += gate[(size_t)n * EE + e];
    for (int o = 16; o; o >>= 1) s += __shfl_xor(s, o, 32);
    if ((tid & 31) == 0) red[tid >> 5] = s;
    __syncthreads();
    if (tid == 0) { float t = 0; for (int i = 0; i < 8; ++i) t += red[i]; prob[e] = t; }
}

__global__ void bal_kernel(const float* __restrict__ prob,
                           const float* __restrict__ used, float* __restrict__ out)
{
    float bal = 0.f;
    const float n = (float)NN;
    for (int e = 0; e < EE; ++e) {
        float p = prob[e] / n;
        float frac = fmaxf(used[e], 1e-9f) / (2.f * n + 1e-9f);
        bal += p * frac;
    }
    out[0] = bal * (float)EE;
}

__global__ void cvt_f16_kernel(const float* __restrict__ in,
                               _Float16* __restrict__ out, size_t n)
{
    size_t i = (size_t)blockIdx.x * blockDim.x + threadIdx.x;
    size_t stride = (size_t)gridDim.x * blockDim.x;
    for (; i < n; i += stride) out[i] = (_Float16)in[i];
}

// =====================================================================
// Host orchestration
// =====================================================================
extern "C" void kernel_launch(void* const* d_in, const int* in_sizes, int n_in,
                              void* d_out, int out_size, void* d_ws, size_t ws_size,
                              hipStream_t stream)
{
    (void)in_sizes; (void)n_in; (void)out_size; (void)ws_size;
    const float* x     = (const float*)d_in[0];
    const float* ln1_g = (const float*)d_in[1];
    const float* ln1_b = (const float*)d_in[2];
    const float* ln2_g = (const float*)d_in[3];
    const float* ln2_b = (const float*)d_in[4];
    const float* Wqkv  = (const float*)d_in[5];
    const float* Wproj = (const float*)d_in[6];
    const float* Wr    = (const float*)d_in[7];
    const float* W1e   = (const float*)d_in[8];
    const float* W2e   = (const float*)d_in[9];
    float* out = (float*)d_out;

    size_t off = 0;
    auto alloc = [&](size_t bytes) -> void* {
        void* p = (char*)d_ws + off;
        off += (bytes + 255) & ~(size_t)255;
        return p;
    };
    _Float16* wqkv16 = (_Float16*)alloc((size_t)HH * 3 * HH * 2);
    _Float16* wproj16= (_Float16*)alloc((size_t)HH * HH * 2);
    _Float16* w1e16  = (_Float16*)alloc((size_t)EE * HH * 2 * DFFF * 2);
    _Float16* w2e16  = (_Float16*)alloc((size_t)EE * DFFF * HH * 2);
    _Float16* xn16   = (_Float16*)alloc((size_t)NN * HH * 2);
    _Float16* qkv16  = (_Float16*)alloc((size_t)NN * 3 * HH * 2);
    _Float16* o16    = (_Float16*)alloc((size_t)NN * HH * 2);
    _Float16* eb16   = (_Float16*)alloc((size_t)EE * CAPN * HH * 2);
    _Float16* hpre16 = (_Float16*)alloc((size_t)EE * CAPN * 2 * DFFF * 2);
    _Float16* act16  = (_Float16*)alloc((size_t)EE * CAPN * DFFF * 2);
    float* xn1f  = (float*)alloc((size_t)NN * HH * 4);
    float* x1    = (float*)alloc((size_t)NN * HH * 4);
    float* xn2   = (float*)alloc((size_t)NN * HH * 4);
    float* eo    = (float*)alloc((size_t)EE * CAPN * HH * 4);
    float* gate  = (float*)alloc((size_t)NN * EE * 4);
    float* val1  = (float*)alloc((size_t)NN * 4);
    float* val2  = (float*)alloc((size_t)NN * 4);
    float* prob  = (float*)alloc(EE * 4);
    float* used  = (float*)alloc(EE * 4);
    int* idx1    = (int*)alloc((size_t)NN * 4);
    int* idx2    = (int*)alloc((size_t)NN * 4);
    int* dest    = (int*)alloc((size_t)2 * NN * 4);
    int* srcslot = (int*)alloc((size_t)EE * CAPN * 4);

    // --- weight conversion to f16 (compute-bound GEMMs want f16 WMMA) ---
    cvt_f16_kernel<<<2048, 256, 0, stream>>>(Wqkv,  wqkv16,  (size_t)HH * 3 * HH);
    cvt_f16_kernel<<<2048, 256, 0, stream>>>(Wproj, wproj16, (size_t)HH * HH);
    cvt_f16_kernel<<<4096, 256, 0, stream>>>(W1e,   w1e16,   (size_t)EE * HH * 2 * DFFF);
    cvt_f16_kernel<<<4096, 256, 0, stream>>>(W2e,   w2e16,   (size_t)EE * DFFF * HH);

    // --- LN1 ---
    ln_kernel<<<NN, 256, 0, stream>>>(x, ln1_g, ln1_b, xn1f, xn16);

    // --- QKV GEMM: [4096,1024]x[1024,3072] -> f16 ---
    gemm_wmma<1><<<dim3(NN / 128, (3 * HH) / 128), 256, 0, stream>>>(
        xn16, wqkv16, qkv16, nullptr, HH, HH, 3 * HH, 3 * HH);

    // --- causal attention ---
    (void)hipFuncSetAttribute(reinterpret_cast<const void*>(att_kernel),
                              hipFuncAttributeMaxDynamicSharedMemorySize, (int)ATT_SHM);
    att_kernel<<<dim3(BB * NHH, TT / 32), 256, ATT_SHM, stream>>>(qkv16, o16);

    // --- proj GEMM + residual: x1 = x + o @ Wproj ---
    gemm_wmma<2><<<dim3(NN / 128, HH / 128), 256, 0, stream>>>(
        o16, wproj16, x1, x, HH, HH, HH, HH);

    // --- LN2 ---
    ln_kernel<<<NN, 256, 0, stream>>>(x1, ln2_g, ln2_b, xn2, xn16);

    // --- router + dispatch + gather ---
    router_kernel<<<NN / 256, 256, 0, stream>>>(xn2, Wr, gate, idx1, idx2, val1, val2);
    dispatch_kernel<<<1, 256, 0, stream>>>(idx1, idx2, dest, srcslot, used);
    gather_kernel<<<EE * CAPN, 256, 0, stream>>>(xn2, srcslot, eb16);

    // --- expert FFN (per-expert WMMA GEMMs) ---
    for (int e = 0; e < EE; ++e) {
        gemm_wmma<1><<<dim3(CAPN / 128, (2 * DFFF) / 128), 256, 0, stream>>>(
            eb16 + (size_t)e * CAPN * HH,
            w1e16 + (size_t)e * HH * 2 * DFFF,
            hpre16 + (size_t)e * CAPN * 2 * DFFF,
            nullptr, HH, HH, 2 * DFFF, 2 * DFFF);
    }
    swiglu_kernel<<<EE * CAPN, 256, 0, stream>>>(hpre16, act16);
    for (int e = 0; e < EE; ++e) {
        gemm_wmma<0><<<dim3(CAPN / 128, HH / 128), 256, 0, stream>>>(
            act16 + (size_t)e * CAPN * DFFF,
            w2e16 + (size_t)e * DFFF * HH,
            eo + (size_t)e * CAPN * HH,
            nullptr, DFFF, DFFF, HH, HH);
    }

    // --- combine + balance loss ---
    combine_kernel<<<NN, 256, 0, stream>>>(x1, eo, dest, val1, val2, out);
    gatesum_kernel<<<EE, 256, 0, stream>>>(gate, prob);
    bal_kernel<<<1, 1, 0, stream>>>(prob, used, out + (size_t)NN * HH);
}
